// GATClassifier_5677946765452
// MI455X (gfx1250) — matrix operand order repaired and checked by reading, not compiled, not run on hardware
//
#include <hip/hip_runtime.h>
#include <stdint.h>

#define IN_C 128
#define HID 64
#define HEADS 4
#define NEG_SLOPE 0.2f
#define BN_EPS 1e-5f

typedef __attribute__((ext_vector_type(16))) __bf16 bf16x16;
typedef __attribute__((ext_vector_type(8)))  float  v8f;

// ---------- helpers ----------
__device__ __forceinline__ unsigned short f2bf(float f) {
  union { float f; unsigned int u; } x; x.f = f;
  unsigned int u = x.u;
  unsigned int r = (u + 0x7FFFu + ((u >> 16) & 1u)) >> 16;  // RNE
  return (unsigned short)r;
}
__device__ __forceinline__ float lrelu(float x) { return x > 0.f ? x : NEG_SLOPE * x; }
__device__ __forceinline__ float elu(float x)   { return x > 0.f ? x : expm1f(x); }

// float atomic max via int/uint punning (valid with init = any real value)
__device__ __forceinline__ void atomicMaxF(float* addr, float val) {
  if (val >= 0.f) atomicMax((int*)addr, __float_as_int(val));
  else            atomicMin((unsigned int*)addr, __float_as_uint(val));
}

// ---------- conversion kernels ----------
__global__ void k_f32_to_bf16(const float* __restrict__ src, unsigned short* __restrict__ dst, int n) {
  int i = blockIdx.x * blockDim.x + threadIdx.x;
  if (i < n) dst[i] = f2bf(src[i]);
}

// W [K,O] fp32 -> Wt [O,K] bf16
__global__ void k_transpose_bf16(const float* __restrict__ W, unsigned short* __restrict__ Wt, int K, int O) {
  int i = blockIdx.x * blockDim.x + threadIdx.x;
  if (i >= K * O) return;
  int o = i / K, k = i % K;
  Wt[i] = f2bf(W[(size_t)k * O + o]);
}

// ---------- WMMA GEMM: C[M,Ncols] = A[M,K](bf16) * Bt[Ncols,K]^T(bf16), fp32 acc ----------
// one 16x16 output tile per wave32; K multiple of 32; M,Ncols multiples of 16
__global__ void k_wmma_gemm(const unsigned short* __restrict__ A,
                            const unsigned short* __restrict__ Bt,
                            float* __restrict__ C,
                            int M, int K, int Ncols,
                            const float* __restrict__ bias, int act) {
  int wave = (blockIdx.x * blockDim.x + threadIdx.x) >> 5;
  int lane = threadIdx.x & 31;
  int tiles_n = Ncols >> 4;
  int tiles_m = M >> 4;
  int tile_m = wave / tiles_n;
  int tile_n = wave - tile_m * tiles_n;
  if (tile_m >= tiles_m) return;  // wave-uniform: EXEC stays all-ones for WMMA

  int row  = lane & 15;   // A-row within tile / Bt-row (= output column) within tile
  int half = lane >> 4;

  const unsigned short* arow = A  + (size_t)(tile_m * 16 + row) * K;
  const unsigned short* brow = Bt + (size_t)(tile_n * 16 + row) * K;

  v8f acc = {};
  for (int k0 = 0; k0 < K; k0 += 32) {
    union { uint4 q[2]; bf16x16 v; } af, bfm;
    // A 16x32 bf16 fragment: lane half h holds K = [8h..8h+7] and [8h+16..8h+23]
    af.q[0] = *reinterpret_cast<const uint4*>(arow + k0 + half * 8);
    af.q[1] = *reinterpret_cast<const uint4*>(arow + k0 + half * 8 + 16);
    // B 32x16 bf16 fragment: lane half h holds K = [8h..8h+7] and [8h+16..8h+23]
    const uint4* bp = reinterpret_cast<const uint4*>(brow + k0 + half * 8);
    bfm.q[0] = bp[0];
    bfm.q[1] = bp[2];
    acc = __builtin_amdgcn_wmma_f32_16x16x32_bf16(false, af.v, false, bfm.v,
                                                  (short)0, acc, false, false);
  }

  // D layout: element j -> row (j + 8*half), col row(lane&15)
  float* crow = C + (size_t)(tile_m * 16 + half * 8) * Ncols + tile_n * 16 + row;
  float b = bias ? bias[tile_n * 16 + row] : 0.f;
#pragma unroll
  for (int j = 0; j < 8; ++j) {
    float v = acc[j] + b;
    if (act) v = elu(v);
    crow[(size_t)j * Ncols] = v;
  }
}

// ---------- attention coefficients: als/ald [N,HEADS] ----------
__global__ void k_attn(const float* __restrict__ hlin, const float* __restrict__ as_,
                       const float* __restrict__ ad_, float* __restrict__ als,
                       float* __restrict__ ald, int n, int ph) {
  int i = blockIdx.x * blockDim.x + threadIdx.x;
  if (i >= n * HEADS) return;
  int node = i >> 2, h = i & 3;
  const float* hp = hlin + (size_t)node * HEADS * ph + h * ph;
  float s = 0.f, d = 0.f;
  for (int c = 0; c < ph; ++c) { float v = hp[c]; s += v * as_[h * ph + c]; d += v * ad_[h * ph + c]; }
  als[i] = s; ald[i] = d;
}

// ---------- segment softmax: self-loop init + edge atomics ----------
__global__ void k_self_max(const float* __restrict__ als, const float* __restrict__ ald,
                           float* __restrict__ mx, int n4) {
  int i = blockIdx.x * blockDim.x + threadIdx.x;
  if (i < n4) mx[i] = lrelu(als[i] + ald[i]);
}
__global__ void k_edge_max(const int* __restrict__ src, const int* __restrict__ dst,
                           const float* __restrict__ als, const float* __restrict__ ald,
                           float* __restrict__ mx, int e4) {
  int t = blockIdx.x * blockDim.x + threadIdx.x;
  if (t >= e4) return;
  int e = t >> 2, h = t & 3;
  int s = src[e], d = dst[e];
  atomicMaxF(&mx[d * HEADS + h], lrelu(als[s * HEADS + h] + ald[d * HEADS + h]));
}
__global__ void k_self_den(const float* __restrict__ als, const float* __restrict__ ald,
                           const float* __restrict__ mx, float* __restrict__ den, int n4) {
  int i = blockIdx.x * blockDim.x + threadIdx.x;
  if (i < n4) den[i] = __expf(lrelu(als[i] + ald[i]) - mx[i]);
}
__global__ void k_edge_den(const int* __restrict__ src, const int* __restrict__ dst,
                           const float* __restrict__ als, const float* __restrict__ ald,
                           const float* __restrict__ mx, float* __restrict__ den, int e4) {
  int t = blockIdx.x * blockDim.x + threadIdx.x;
  if (t >= e4) return;
  int e = t >> 2, h = t & 3;
  int s = src[e], d = dst[e];
  float ev = lrelu(als[s * HEADS + h] + ald[d * HEADS + h]);
  atomicAdd(&den[d * HEADS + h], __expf(ev - mx[d * HEADS + h]));
}

// ---------- aggregation, concat layers (ph=16, out width HID) ----------
__global__ void k_self_agg_cat(const float* __restrict__ hlin, const float* __restrict__ als,
                               const float* __restrict__ ald, const float* __restrict__ mx,
                               const float* __restrict__ den, float* __restrict__ out,
                               int n, int ph) {
  int i = blockIdx.x * blockDim.x + threadIdx.x;
  if (i >= n * HID) return;
  int node = i / HID, j = i - node * HID, h = j / ph;
  int nh = node * HEADS + h;
  float a = __expf(lrelu(als[nh] + ald[nh]) - mx[nh]) / den[nh];
  out[i] = hlin[i] * a;
}
__global__ void k_edge_agg_cat(const int* __restrict__ src, const int* __restrict__ dst,
                               const float* __restrict__ hlin, const float* __restrict__ als,
                               const float* __restrict__ ald, const float* __restrict__ mx,
                               const float* __restrict__ den, float* __restrict__ out,
                               int E, int ph) {
  int t = blockIdx.x * blockDim.x + threadIdx.x;
  if (t >= E * HID) return;
  int e = t / HID, j = t - e * HID, h = j / ph;
  int s = src[e], d = dst[e];
  int dh = d * HEADS + h;
  float ev = lrelu(als[s * HEADS + h] + ald[dh]);
  float a = __expf(ev - mx[dh]) / den[dh];
  atomicAdd(&out[(size_t)d * HID + j], hlin[(size_t)s * HID + j] * a);
}

// ---------- aggregation, mean layer (ph=HID, hlin row = HEADS*HID, out width HID) ----------
__global__ void k_self_agg_mean(const float* __restrict__ hlin, const float* __restrict__ als,
                                const float* __restrict__ ald, const float* __restrict__ mx,
                                const float* __restrict__ den, float* __restrict__ out, int n) {
  int i = blockIdx.x * blockDim.x + threadIdx.x;
  if (i >= n * HID) return;
  int node = i / HID, j = i - node * HID;
  float acc = 0.f;
#pragma unroll
  for (int h = 0; h < HEADS; ++h) {
    int nh = node * HEADS + h;
    float a = __expf(lrelu(als[nh] + ald[nh]) - mx[nh]) / den[nh];
    acc += hlin[(size_t)node * (HEADS * HID) + h * HID + j] * a;
  }
  out[i] = 0.25f * acc;
}
__global__ void k_edge_agg_mean(const int* __restrict__ src, const int* __restrict__ dst,
                                const float* __restrict__ hlin, const float* __restrict__ als,
                                const float* __restrict__ ald, const float* __restrict__ mx,
                                const float* __restrict__ den, float* __restrict__ out, int E) {
  int t = blockIdx.x * blockDim.x + threadIdx.x;
  if (t >= E * HID) return;
  int e = t / HID, j = t - e * HID;
  int s = src[e], d = dst[e];
  float acc = 0.f;
#pragma unroll
  for (int h = 0; h < HEADS; ++h) {
    int dh = d * HEADS + h;
    float ev = lrelu(als[s * HEADS + h] + ald[dh]);
    float a = __expf(ev - mx[dh]) / den[dh];
    acc += hlin[(size_t)s * (HEADS * HID) + h * HID + j] * a;
  }
  atomicAdd(&out[(size_t)d * HID + j], 0.25f * acc);
}

// ---------- (x + layer_bias) -> BN -> optional ELU -> bf16 ----------
__global__ void k_bn(const float* __restrict__ in, const float* __restrict__ lb,
                     const float* __restrict__ g, const float* __restrict__ bb,
                     const float* __restrict__ m, const float* __restrict__ v,
                     unsigned short* __restrict__ outbf, int n, int doelu) {
  int i = blockIdx.x * blockDim.x + threadIdx.x;
  if (i >= n * HID) return;
  int c = i % HID;
  float x = in[i] + lb[c];
  float y = g[c] * (x - m[c]) * rsqrtf(v[c] + BN_EPS) + bb[c];
  if (doelu) y = elu(y);
  outbf[i] = f2bf(y);
}

// ---------- final 32 -> 1 ----------
__global__ void k_final(const float* __restrict__ hid, const float* __restrict__ cW2,
                        const float* __restrict__ cb2, float* __restrict__ out, int n) {
  int i = blockIdx.x * blockDim.x + threadIdx.x;
  if (i >= n) return;
  float acc = cb2[0];
  const float* hp = hid + (size_t)i * 32;
#pragma unroll
  for (int c = 0; c < 32; ++c) acc += hp[c] * cW2[c];
  out[i] = acc;
}

static inline int gsz(long long n, int b) { return (int)((n + b - 1) / b); }

extern "C" void kernel_launch(void* const* d_in, const int* in_sizes, int n_in,
                              void* d_out, int out_size, void* d_ws, size_t ws_size,
                              hipStream_t stream) {
  const float* x     = (const float*)d_in[0];
  const int*   eidx  = (const int*)d_in[1];
  const float* W1    = (const float*)d_in[2];
  const float* a1s   = (const float*)d_in[3];
  const float* a1d   = (const float*)d_in[4];
  const float* b1    = (const float*)d_in[5];
  const float* W2    = (const float*)d_in[6];
  const float* a2s   = (const float*)d_in[7];
  const float* a2d   = (const float*)d_in[8];
  const float* b2    = (const float*)d_in[9];
  const float* W3    = (const float*)d_in[10];
  const float* a3s   = (const float*)d_in[11];
  const float* a3d   = (const float*)d_in[12];
  const float* b3    = (const float*)d_in[13];
  const float* bn1g  = (const float*)d_in[14];
  const float* bn1b  = (const float*)d_in[15];
  const float* bn1m  = (const float*)d_in[16];
  const float* bn1v  = (const float*)d_in[17];
  const float* bn2g  = (const float*)d_in[18];
  const float* bn2b  = (const float*)d_in[19];
  const float* bn2m  = (const float*)d_in[20];
  const float* bn2v  = (const float*)d_in[21];
  const float* bn3g  = (const float*)d_in[22];
  const float* bn3b  = (const float*)d_in[23];
  const float* bn3m  = (const float*)d_in[24];
  const float* bn3v  = (const float*)d_in[25];
  const float* cW1   = (const float*)d_in[26];
  const float* cb1   = (const float*)d_in[27];
  const float* cW2   = (const float*)d_in[28];
  const float* cb2   = (const float*)d_in[29];

  const int N = in_sizes[0] / IN_C;
  const int E = in_sizes[1] / 2;
  const int* src = eidx;
  const int* dst = eidx + E;
  float* out = (float*)d_out;

  // workspace layout (256B aligned slices)
  size_t off = 0;
  auto walloc = [&](size_t bytes) -> void* {
    void* p = (char*)d_ws + off;
    off += (bytes + 255) & ~(size_t)255;
    return p;
  };
  unsigned short* xb   = (unsigned short*)walloc((size_t)N * IN_C * 2);
  unsigned short* w1t  = (unsigned short*)walloc((size_t)HID * IN_C * 2);       // [64,128]
  unsigned short* w2t  = (unsigned short*)walloc((size_t)HID * HID * 2);        // [64,64]
  unsigned short* w3t  = (unsigned short*)walloc((size_t)HEADS * HID * HID * 2);// [256,64]
  unsigned short* cw1t = (unsigned short*)walloc((size_t)32 * HID * 2);         // [32,64]
  float* hlin = (float*)walloc((size_t)N * HEADS * HID * 4);                    // up to [N,256]
  float* als  = (float*)walloc((size_t)N * HEADS * 4);
  float* ald  = (float*)walloc((size_t)N * HEADS * 4);
  float* mx   = (float*)walloc((size_t)N * HEADS * 4);
  float* den  = (float*)walloc((size_t)N * HEADS * 4);
  float* agg  = (float*)walloc((size_t)N * HID * 4);
  unsigned short* hbf = (unsigned short*)walloc((size_t)N * HID * 2);
  float* chid = (float*)walloc((size_t)N * 32 * 4);
  (void)ws_size; (void)n_in; (void)out_size;

  const int B = 256;
  const int n4 = N * HEADS, e4 = E * HEADS;
  const long long nH = (long long)N * HID, eH = (long long)E * HID;

  // conversions
  k_f32_to_bf16<<<gsz((long long)N * IN_C, B), B, 0, stream>>>(x, xb, N * IN_C);
  k_transpose_bf16<<<gsz(IN_C * HID, B), B, 0, stream>>>(W1, w1t, IN_C, HID);
  k_transpose_bf16<<<gsz(HID * HID, B), B, 0, stream>>>(W2, w2t, HID, HID);
  k_transpose_bf16<<<gsz(HID * HEADS * HID, B), B, 0, stream>>>(W3, w3t, HID, HEADS * HID);
  k_transpose_bf16<<<gsz(HID * 32, B), B, 0, stream>>>(cW1, cw1t, HID, 32);

  auto gemm = [&](const unsigned short* A, const unsigned short* Bt, float* C,
                  int M, int K, int Nc, const float* bias, int act) {
    long long waves = (long long)(M / 16) * (Nc / 16);
    k_wmma_gemm<<<gsz(waves * 32, B), B, 0, stream>>>(A, Bt, C, M, K, Nc, bias, act);
  };

  // ---- Layer 1 (concat, ph=16) ----
  gemm(xb, w1t, hlin, N, IN_C, HID, nullptr, 0);
  k_attn<<<gsz(n4, B), B, 0, stream>>>(hlin, a1s, a1d, als, ald, N, 16);
  k_self_max<<<gsz(n4, B), B, 0, stream>>>(als, ald, mx, n4);
  k_edge_max<<<gsz(e4, B), B, 0, stream>>>(src, dst, als, ald, mx, e4);
  k_self_den<<<gsz(n4, B), B, 0, stream>>>(als, ald, mx, den, n4);
  k_edge_den<<<gsz(e4, B), B, 0, stream>>>(src, dst, als, ald, mx, den, e4);
  k_self_agg_cat<<<gsz(nH, B), B, 0, stream>>>(hlin, als, ald, mx, den, agg, N, 16);
  k_edge_agg_cat<<<gsz(eH, B), B, 0, stream>>>(src, dst, hlin, als, ald, mx, den, agg, E, 16);
  k_bn<<<gsz(nH, B), B, 0, stream>>>(agg, b1, bn1g, bn1b, bn1m, bn1v, hbf, N, 1);

  // ---- Layer 2 (concat, ph=16) ----
  gemm(hbf, w2t, hlin, N, HID, HID, nullptr, 0);
  k_attn<<<gsz(n4, B), B, 0, stream>>>(hlin, a2s, a2d, als, ald, N, 16);
  k_self_max<<<gsz(n4, B), B, 0, stream>>>(als, ald, mx, n4);
  k_edge_max<<<gsz(e4, B), B, 0, stream>>>(src, dst, als, ald, mx, e4);
  k_self_den<<<gsz(n4, B), B, 0, stream>>>(als, ald, mx, den, n4);
  k_edge_den<<<gsz(e4, B), B, 0, stream>>>(src, dst, als, ald, mx, den, e4);
  k_self_agg_cat<<<gsz(nH, B), B, 0, stream>>>(hlin, als, ald, mx, den, agg, N, 16);
  k_edge_agg_cat<<<gsz(eH, B), B, 0, stream>>>(src, dst, hlin, als, ald, mx, den, agg, E, 16);
  k_bn<<<gsz(nH, B), B, 0, stream>>>(agg, b2, bn2g, bn2b, bn2m, bn2v, hbf, N, 1);

  // ---- Layer 3 (mean over heads, ph=64) ----
  gemm(hbf, w3t, hlin, N, HID, HEADS * HID, nullptr, 0);
  k_attn<<<gsz(n4, B), B, 0, stream>>>(hlin, a3s, a3d, als, ald, N, HID);
  k_self_max<<<gsz(n4, B), B, 0, stream>>>(als, ald, mx, n4);
  k_edge_max<<<gsz(e4, B), B, 0, stream>>>(src, dst, als, ald, mx, e4);
  k_self_den<<<gsz(n4, B), B, 0, stream>>>(als, ald, mx, den, n4);
  k_edge_den<<<gsz(e4, B), B, 0, stream>>>(src, dst, als, ald, mx, den, e4);
  k_self_agg_mean<<<gsz(nH, B), B, 0, stream>>>(hlin, als, ald, mx, den, agg, N);
  k_edge_agg_mean<<<gsz(eH, B), B, 0, stream>>>(src, dst, hlin, als, ald, mx, den, agg, E);
  k_bn<<<gsz(nH, B), B, 0, stream>>>(agg, b3, bn3g, bn3b, bn3m, bn3v, hbf, N, 0); // no ELU

  // ---- classifier: 64 -> 32 (bias+ELU fused) -> 1 ----
  gemm(hbf, cw1t, chid, N, HID, 32, cb1, 1);
  k_final<<<gsz(N, B), B, 0, stream>>>(chid, cW2, cb2, out, N);
}
